// POCACritic_2534030705143
// MI455X (gfx1250) — compile-verified, matrix operand order checked by reading, not run on hardware
//
#include <hip/hip_runtime.h>

typedef __attribute__((ext_vector_type(16))) _Float16 v16h;
typedef __attribute__((ext_vector_type(8)))  float    v8f;

#define B_TOT  8192
#define NAG    32
#define OBSD   128
#define H      256
#define HDIM   64
#define LN_EPS 1e-5f

// packed-weight offsets in d_ws (units: halves)
#define OFF_EMB 0        // 4 ktiles  -> 32768 halves
#define OFF_Q   32768    // 8 ktiles  -> 65536 halves each
#define OFF_K   98304
#define OFF_V   163840
#define OFF_O   229376

#define WMMA(a,b,c) __builtin_amdgcn_wmma_f32_16x16x32_f16(false,(a),false,(b),(short)0,(c),false,false)

__device__ __forceinline__ float siluf(float x) { return x / (1.0f + __expf(-x)); }

// 16x32 f16 A-operand tile from row-major f16 memory (also valid as B-operand of a transposed matrix).
// lane<16 : row m0+lane, K = {0..7, 16..23};  lane>=16 : row m0+lane-16, K = {8..15, 24..31}
__device__ __forceinline__ v16h load_a16(const _Float16* base, int stride, int m0, int k0, int lane) {
  const int half = lane >> 4;
  const _Float16* p = base + (m0 + (lane & 15)) * stride + k0 + half * 8;
  v16h a;
  *(uint4*)&a       = *(const uint4*)(p);
  *((uint4*)&a + 1) = *(const uint4*)(p + 16);
  return a;
}

// Same tile shape, but source is row-major f32 in global memory (obs) -> convert to f16.
__device__ __forceinline__ v16h load_a32(const float* base, int stride, int m0, int k0, int lane) {
  const int half = lane >> 4;
  const float* p = base + (m0 + (lane & 15)) * stride + k0 + half * 8;
  v16h a;
#pragma unroll
  for (int i = 0; i < 8; ++i) a[i] = (_Float16)p[i];
#pragma unroll
  for (int i = 0; i < 8; ++i) a[8 + i] = (_Float16)p[16 + i];
  return a;
}

// Pre-packed B-operand tile: [ktile][ntile][lane][16 halves]
__device__ __forceinline__ v16h load_bpk2(const _Float16* wp, int kt, int tn, int lane) {
  return *(const v16h*)(wp + ((size_t)(((kt << 4) + tn) << 5) + lane) * 16);
}

// 32xK (K=32*ksteps) x 32 GEMM: A from LDS f16 (row-major), B from packed weights.
__device__ __forceinline__ void mm_lds(const _Float16* A, int strideA, const _Float16* Wp,
                                       int c0, int ksteps, int lane, v8f acc[2][2]) {
  const int tn0 = c0 >> 4;
  for (int kt = 0; kt < ksteps; ++kt) {
    v16h a0 = load_a16(A, strideA, 0,  kt * 32, lane);
    v16h a1 = load_a16(A, strideA, 16, kt * 32, lane);
    v16h b0 = load_bpk2(Wp, kt, tn0,     lane);
    v16h b1 = load_bpk2(Wp, kt, tn0 + 1, lane);
    acc[0][0] = WMMA(a0, b0, acc[0][0]);
    acc[0][1] = WMMA(a0, b1, acc[0][1]);
    acc[1][0] = WMMA(a1, b0, acc[1][0]);
    acc[1][1] = WMMA(a1, b1, acc[1][1]);
  }
}

// LayerNorm over last dim of a 32x256 f16 LDS buffer, in place. red = 512-float scratch.
__device__ void layernorm32x256(_Float16* buf, float* red, int tid) {
  const int row = tid >> 3, j = tid & 7;
  {
    const _Float16* p = buf + row * H + j * 32;
    float s = 0.f, ss = 0.f;
#pragma unroll 8
    for (int i = 0; i < 32; ++i) { float v = (float)p[i]; s += v; ss += v * v; }
    red[tid] = s; red[256 + tid] = ss;
  }
  __syncthreads();
  float s = 0.f, ss = 0.f;
#pragma unroll
  for (int i = 0; i < 8; ++i) { s += red[row * 8 + i]; ss += red[256 + row * 8 + i]; }
  const float m   = s * (1.0f / H);
  const float inv = rsqrtf(ss * (1.0f / H) - m * m + LN_EPS);
  _Float16* q = buf + row * H + j * 32;
#pragma unroll 8
  for (int i = 0; i < 32; ++i) { float v = (float)q[i]; q[i] = (_Float16)((v - m) * inv); }
  __syncthreads();
}

// Convert f32 weight (Kx256 row-major) into per-lane WMMA-B packed f16 layout.
__global__ __launch_bounds__(256) void pack_w_kernel(const float* __restrict__ src,
                                                     _Float16* __restrict__ dst, int ktiles) {
  const int i = blockIdx.x * 256 + threadIdx.x;
  const int total = ktiles * 8192;             // 16 ntiles * 32 lanes * 16 halves per ktile
  if (i >= total) return;
  const int e = i & 15, lane = (i >> 4) & 31, tn = (i >> 9) & 15, tk = i >> 13;
  const int kin = e + ((e >= 8) ? 8 : 0) + ((lane >= 16) ? 8 : 0);
  const int n = tn * 16 + (lane & 15);
  dst[i] = (_Float16)src[(tk * 32 + kin) * 256 + n];
}

__global__ __launch_bounds__(256)
void poca_fused(const float* __restrict__ obs, const _Float16* __restrict__ wpk,
                const float* __restrict__ b_emb,
                const float* __restrict__ bq, const float* __restrict__ bk,
                const float* __restrict__ bv, const float* __restrict__ bo,
                const float* __restrict__ W1, const float* __restrict__ b1,
                const float* __restrict__ W2, const float* __restrict__ b2,
                const float* __restrict__ Wval, const float* __restrict__ bval,
                float* __restrict__ out) {
  __shared__ __align__(32) unsigned char smem[65536];
  _Float16* xb = (_Float16*)(smem);            // A: x (32x256 f16) -> later pooled/MLP floats
  _Float16* qb = (_Float16*)(smem + 16384);    // B: LN scratch -> q -> vT (256x32)
  _Float16* kb = (_Float16*)(smem + 32768);    // C: k -> attn_out -> LN scratch
  _Float16* db = (_Float16*)(smem + 49152);    // D: attn probs (4x32x32) -> y (32x256)

  const int b    = blockIdx.x;
  const int tid  = threadIdx.x;
  const int wave = tid >> 5;
  const int lane = tid & 31;
  const int c0   = wave * 32;                  // this wave's H-column slice
  const int half = lane >> 4, ln = lane & 15;

  const float* obsb = obs + (size_t)b * NAG * OBSD;

  // ---- Phase 1: ent = silu(obs @ W_emb + b_emb) ----
  {
    v8f acc[2][2] = {};
    const _Float16* wemb = wpk + OFF_EMB;
    const int tn0 = c0 >> 4;
    for (int kt = 0; kt < 4; ++kt) {
      v16h a0 = load_a32(obsb, OBSD, 0,  kt * 32, lane);
      v16h a1 = load_a32(obsb, OBSD, 16, kt * 32, lane);
      v16h b0 = load_bpk2(wemb, kt, tn0,     lane);
      v16h b1 = load_bpk2(wemb, kt, tn0 + 1, lane);
      acc[0][0] = WMMA(a0, b0, acc[0][0]);
      acc[0][1] = WMMA(a0, b1, acc[0][1]);
      acc[1][0] = WMMA(a1, b0, acc[1][0]);
      acc[1][1] = WMMA(a1, b1, acc[1][1]);
    }
#pragma unroll
    for (int nt = 0; nt < 2; ++nt) {
      const int col = c0 + nt * 16 + ln;
      const float bias = b_emb[col];
#pragma unroll
      for (int mt = 0; mt < 2; ++mt)
#pragma unroll
        for (int r = 0; r < 8; ++r) {
          const int m = mt * 16 + r + 8 * half;
          xb[m * H + col] = (_Float16)siluf(acc[mt][nt][r] + bias);
        }
    }
  }
  __syncthreads();
  layernorm32x256(xb, (float*)qb, tid);        // x = LN(ent)

  // ---- Phase 2a: q, k projections ----
  {
    v8f accq[2][2] = {}, acck[2][2] = {};
    mm_lds(xb, H, wpk + OFF_Q, c0, 8, lane, accq);
    mm_lds(xb, H, wpk + OFF_K, c0, 8, lane, acck);
#pragma unroll
    for (int nt = 0; nt < 2; ++nt) {
      const int col = c0 + nt * 16 + ln;
      const float biq = bq[col], bik = bk[col];
#pragma unroll
      for (int mt = 0; mt < 2; ++mt)
#pragma unroll
        for (int r = 0; r < 8; ++r) {
          const int m = mt * 16 + r + 8 * half;
          qb[m * H + col] = (_Float16)(accq[mt][nt][r] + biq);
          kb[m * H + col] = (_Float16)(acck[mt][nt][r] + bik);
        }
    }
  }
  __syncthreads();

  // ---- Phase 3: scores = softmax(q k^T / 8), per head (2 waves/head) ----
  {
    const int h = wave >> 1, mt = wave & 1;
    v8f s0 = {}, s1 = {};
    for (int kk = 0; kk < 2; ++kk) {
      v16h a   = load_a16(qb, H, mt * 16, h * HDIM + kk * 32, lane);
      v16h bt0 = load_a16(kb, H, 0,       h * HDIM + kk * 32, lane); // k rows == B of k^T
      v16h bt1 = load_a16(kb, H, 16,      h * HDIM + kk * 32, lane);
      s0 = WMMA(a, bt0, s0);
      s1 = WMMA(a, bt1, s1);
    }
    _Float16* ap = db + h * (NAG * NAG);
#pragma unroll
    for (int r = 0; r < 8; ++r) {
      float e0 = s0[r] * 0.125f, e1 = s1[r] * 0.125f;
      float mx = fmaxf(e0, e1);
#pragma unroll
      for (int msk = 8; msk >= 1; msk >>= 1) mx = fmaxf(mx, __shfl_xor(mx, msk, 32));
      e0 = __expf(e0 - mx); e1 = __expf(e1 - mx);
      float sum = e0 + e1;
#pragma unroll
      for (int msk = 8; msk >= 1; msk >>= 1) sum += __shfl_xor(sum, msk, 32);
      const float inv = 1.0f / sum;
      const int m = mt * 16 + r + 8 * half;
      ap[m * NAG + ln]      = (_Float16)(e0 * inv);
      ap[m * NAG + 16 + ln] = (_Float16)(e1 * inv);
    }
  }
  __syncthreads();

  // ---- Phase 2b: v projection, stored transposed vT[col][agent] (reuses q region) ----
  {
    v8f acc[2][2] = {};
    mm_lds(xb, H, wpk + OFF_V, c0, 8, lane, acc);
    _Float16* vt = qb;                          // vT: 256 x 32 f16
#pragma unroll
    for (int nt = 0; nt < 2; ++nt) {
      const int col = c0 + nt * 16 + ln;
      const float biv = bv[col];
#pragma unroll
      for (int mt = 0; mt < 2; ++mt)
#pragma unroll
        for (int r = 0; r < 8; ++r) {
          const int m = mt * 16 + r + 8 * half;
          vt[col * NAG + m] = (_Float16)(acc[mt][nt][r] + biv);
        }
    }
  }
  __syncthreads();

  // ---- Phase 4: attn_out = probs @ v (writes over k region) ----
  {
    const int h = wave >> 1, mt = wave & 1;
    const _Float16* ap = db + h * (NAG * NAG);
    const _Float16* vt = qb;
    v16h a = load_a16(ap, NAG, mt * 16, 0, lane);
#pragma unroll
    for (int j = 0; j < 4; ++j) {
      v16h bt = load_a16(vt, NAG, h * HDIM + j * 16, 0, lane); // vT rows -> B operand of v
      v8f d = {};
      d = WMMA(a, bt, d);
#pragma unroll
      for (int r = 0; r < 8; ++r) {
        const int m = mt * 16 + r + 8 * half;
        kb[m * H + h * HDIM + j * 16 + ln] = (_Float16)d[r];
      }
    }
  }
  __syncthreads();

  // ---- Phase 5: y = LN(attn_out @ Wo + bo + x) ----
  {
    v8f acc[2][2] = {};
    mm_lds(kb, H, wpk + OFF_O, c0, 8, lane, acc);
#pragma unroll
    for (int nt = 0; nt < 2; ++nt) {
      const int col = c0 + nt * 16 + ln;
      const float bio = bo[col];
#pragma unroll
      for (int mt = 0; mt < 2; ++mt)
#pragma unroll
        for (int r = 0; r < 8; ++r) {
          const int m = mt * 16 + r + 8 * half;
          db[m * H + col] = (_Float16)(acc[mt][nt][r] + bio + (float)xb[m * H + col]);
        }
    }
  }
  __syncthreads();
  layernorm32x256(db, (float*)kb, tid);

  // ---- Phase 6: mean pool over agents ----
  float* fsm = (float*)xb;  // [0..255] pooled, [256..511] h1, [512..767] h2, [768] value acc
  {
    float s = 0.f;
#pragma unroll 8
    for (int n = 0; n < NAG; ++n) s += (float)db[n * H + tid];
    fsm[tid] = s * (1.0f / NAG);
  }
  __syncthreads();

  // ---- Phase 7: MLP tail + value head ----
  {
    float a = b1[tid];
    for (int k = 0; k < H; ++k) a += fsm[k] * W1[k * H + tid];
    fsm[256 + tid] = siluf(a);
  }
  __syncthreads();
  {
    float a = b2[tid];
    for (int k = 0; k < H; ++k) a += fsm[256 + k] * W2[k * H + tid];
    fsm[512 + tid] = siluf(a);
  }
  __syncthreads();
  if (tid == 0) fsm[768] = bval[0] + Wval[H];   // norm-agent-count feature == 1.0
  __syncthreads();
  atomicAdd(&fsm[768], fsm[512 + tid] * Wval[tid]);
  __syncthreads();
  if (tid == 0) out[b] = fsm[768];
}

extern "C" void kernel_launch(void* const* d_in, const int* in_sizes, int n_in,
                              void* d_out, int out_size, void* d_ws, size_t ws_size,
                              hipStream_t stream) {
  (void)in_sizes; (void)n_in; (void)out_size; (void)ws_size;
  const float* obs  = (const float*)d_in[0];
  const float* Wemb = (const float*)d_in[1];  const float* bemb = (const float*)d_in[2];
  const float* Wq   = (const float*)d_in[3];  const float* bq   = (const float*)d_in[4];
  const float* Wk   = (const float*)d_in[5];  const float* bk   = (const float*)d_in[6];
  const float* Wv   = (const float*)d_in[7];  const float* bv   = (const float*)d_in[8];
  const float* Wo   = (const float*)d_in[9];  const float* bo   = (const float*)d_in[10];
  const float* W1   = (const float*)d_in[11]; const float* b1   = (const float*)d_in[12];
  const float* W2   = (const float*)d_in[13]; const float* b2   = (const float*)d_in[14];
  const float* Wval = (const float*)d_in[15]; const float* bval = (const float*)d_in[16];
  float* out = (float*)d_out;
  _Float16* wp = (_Float16*)d_ws;

  pack_w_kernel<<<128, 256, 0, stream>>>(Wemb, wp + OFF_EMB, 4);
  pack_w_kernel<<<256, 256, 0, stream>>>(Wq,   wp + OFF_Q,   8);
  pack_w_kernel<<<256, 256, 0, stream>>>(Wk,   wp + OFF_K,   8);
  pack_w_kernel<<<256, 256, 0, stream>>>(Wv,   wp + OFF_V,   8);
  pack_w_kernel<<<256, 256, 0, stream>>>(Wo,   wp + OFF_O,   8);

  poca_fused<<<B_TOT, 256, 0, stream>>>(obs, wp, bemb, bq, bk, bv, bo,
                                        W1, b1, W2, b2, Wval, bval, out);
}
